// IMGNetwork_85512798863738
// MI455X (gfx1250) — matrix operand order repaired and checked by reading, not compiled, not run on hardware
//
#include <hip/hip_runtime.h>
#include <math.h>

typedef __attribute__((ext_vector_type(16))) _Float16 v16h;
typedef __attribute__((ext_vector_type(8)))  float    v8f;

#define NPTS          2097152u
#define BLOCK_T       128
#define PTS_PER_BLOCK 128
#define NBLOCKS       (NPTS / PTS_PER_BLOCK)   // 16384

struct LevelMeta {
    float    scale[16];
    unsigned rs[16];     // row stride = r+1
    unsigned off[16];    // element offset into embedding table
};

__device__ __forceinline__ v16h loadA(const _Float16* row_base, int hi) {
    // row_base points at act[m][chunk*32]; contiguous k order used here (B packed alike): j=0..15 -> k=16*hi+j.
    // Two 16B LDS loads.
    union { v16h h; int4 q[2]; } u;
    const int4* src = (const int4*)(row_base + hi * 16);
    u.q[0] = src[0];
    u.q[1] = src[1];
    return u.h;
}

__device__ __forceinline__ v16h loadB(const _Float16* buf, int grp, int lane) {
    union { v16h h; int4 q[2]; } u;
    const int4* src = (const int4*)(buf + grp * 512 + lane * 16);
    u.q[0] = src[0];
    u.q[1] = src[1];
    return u.h;
}

__launch_bounds__(BLOCK_T)
__global__ void fused_hashgrid_mlp(const float* __restrict__ x,
                                   const float* __restrict__ cmax,
                                   const float* __restrict__ emb,
                                   const float* __restrict__ w0,
                                   const float* __restrict__ b0,
                                   const float* __restrict__ w1,
                                   const float* __restrict__ b1,
                                   const float* __restrict__ w2,
                                   const float* __restrict__ b2,
                                   float* __restrict__ out,
                                   LevelMeta meta)
{
    // Pre-swizzled weight tiles (WMMA 16-bit B layout, 16 contiguous halves/lane)
    __shared__ __align__(16) _Float16 sB0[2048];           // 4 ntiles  x 512
    __shared__ __align__(16) _Float16 sB1[4096];           // 4x2 grps  x 512
    __shared__ __align__(16) _Float16 sB2[1024];           // 2 grps    x 512
    // Per-wave activation buffers: [0..2047]=bufA (act0/act2), [2048..4095]=bufB (act1)
    __shared__ __align__(16) _Float16 sAct[BLOCK_T / 32][4096];

    const int tid  = threadIdx.x;
    const int lane = tid & 31;
    const int wid  = tid >> 5;
    const int n    = lane & 15;
    const int hi   = lane >> 4;

    // ---- fill rearranged weights (f32 -> f16), cooperative across block ----
    for (int e = tid; e < 2048; e += BLOCK_T) {
        int j = e & 15, l = (e >> 4) & 31, nt = e >> 9;
        int k = 16 * (l >> 4) + j;                       // K index (0..31)
        sB0[e] = (_Float16)w0[(nt * 16 + (l & 15)) * 32 + k];
    }
    for (int e = tid; e < 4096; e += BLOCK_T) {
        int j = e & 15, l = (e >> 4) & 31, g = e >> 9;   // g = nt*2 + c
        int k = 32 * (g & 1) + 16 * (l >> 4) + j;        // K index (0..63)
        sB1[e] = (_Float16)w1[((g >> 1) * 16 + (l & 15)) * 64 + k];
    }
    for (int e = tid; e < 1024; e += BLOCK_T) {
        int j = e & 15, l = (e >> 4) & 31, c = e >> 9;
        int nn = l & 15;
        int k = 32 * c + 16 * (l >> 4) + j;
        sB2[e] = (nn < 3) ? (_Float16)w2[nn * 64 + k] : (_Float16)0.0f;
    }

    // ---- hash-grid encode: one point per lane, all levels dense ----
    const unsigned pbase = blockIdx.x * PTS_PER_BLOCK + wid * 32;
    const unsigned p = pbase + lane;
    const float x0 = x[2u * p]     / cmax[1];   // cmax.flip(-1)
    const float x1 = x[2u * p + 1] / cmax[0];

    const float2* E = (const float2*)emb;
    float f[32];
#pragma unroll
    for (int l = 0; l < 16; ++l) {
        const float    s   = meta.scale[l];
        const unsigned rs  = meta.rs[l];
        const unsigned off = meta.off[l];
        float p0 = fmaf(x0, s, 0.5f);
        float p1 = fmaf(x1, s, 0.5f);
        float g0 = floorf(p0), g1 = floorf(p1);
        float fr0 = p0 - g0, fr1 = p1 - g1;
        unsigned u0 = (unsigned)g0, u1 = (unsigned)g1;
        unsigned base = off + u1 * rs + u0;              // idx < size always (dense)
        float2 e00 = E[base];
        float2 e10 = E[base + 1u];
        float2 e01 = E[base + rs];
        float2 e11 = E[base + rs + 1u];
        float wx0 = 1.0f - fr0, wy0 = 1.0f - fr1;
        float w00 = wx0 * wy0, w10 = fr0 * wy0, w01 = wx0 * fr1, w11 = fr0 * fr1;
        float a0 = w00 * e00.x;
        a0 = fmaf(w10, e10.x, a0); a0 = fmaf(w01, e01.x, a0); a0 = fmaf(w11, e11.x, a0);
        float a1 = w00 * e00.y;
        a1 = fmaf(w10, e10.y, a1); a1 = fmaf(w01, e01.y, a1); a1 = fmaf(w11, e11.y, a1);
        f[2 * l]     = a0;
        f[2 * l + 1] = a1;
    }

    // out_lc0 (pre-relu, fp32) at out + N*3 + p*32
    {
        float4* o = (float4*)(out + (size_t)NPTS * 3u + (size_t)p * 32u);
#pragma unroll
        for (int q = 0; q < 8; ++q)
            o[q] = make_float4(f[4 * q], f[4 * q + 1], f[4 * q + 2], f[4 * q + 3]);
    }

    _Float16* bufA = &sAct[wid][0];
    _Float16* bufB = &sAct[wid][2048];

    // relu -> f16, store act0 naturally: bufA[point][feat 0..31]
    {
        union { _Float16 h[32]; int4 q[4]; } pk;
#pragma unroll
        for (int i = 0; i < 32; ++i) pk.h[i] = (_Float16)fmaxf(f[i], 0.0f);
        int4* dst = (int4*)(bufA + lane * 32);
#pragma unroll
        for (int q = 0; q < 4; ++q) dst[q] = pk.q[q];
    }

    __syncthreads();   // weights + activations visible

    const v8f zero = {0.f, 0.f, 0.f, 0.f, 0.f, 0.f, 0.f, 0.f};

    // ---- layer 0: [16x32] x [32x64], 4 n-tiles, 2 m-tiles ----
    v16h A0[2];
#pragma unroll
    for (int mt = 0; mt < 2; ++mt)
        A0[mt] = loadA(bufA + (mt * 16 + n) * 32, hi);

#pragma unroll
    for (int nt = 0; nt < 4; ++nt) {
        v16h B = loadB(sB0, nt, lane);
        float bias = b0[nt * 16 + n];
#pragma unroll
        for (int mt = 0; mt < 2; ++mt) {
            v8f acc = __builtin_amdgcn_wmma_f32_16x16x32_f16(false, A0[mt], false, B,
                                                             (short)0, zero, false, false);
#pragma unroll
            for (int r = 0; r < 8; ++r) {
                float v = fmaxf(acc[r] + bias, 0.0f);
                int pt = mt * 16 + r + 8 * hi;           // C/D layout: m = r + 8*hi
                bufB[pt * 64 + nt * 16 + n] = (_Float16)v;
            }
        }
    }

    // ---- layer 1: [16x64] x [64x64], chain two K=32 WMMAs ----
    v16h A1[2][2];
#pragma unroll
    for (int mt = 0; mt < 2; ++mt)
#pragma unroll
        for (int c = 0; c < 2; ++c)
            A1[mt][c] = loadA(bufB + (mt * 16 + n) * 64 + c * 32, hi);

#pragma unroll
    for (int nt = 0; nt < 4; ++nt) {
        v16h Bc0 = loadB(sB1, nt * 2 + 0, lane);
        v16h Bc1 = loadB(sB1, nt * 2 + 1, lane);
        float bias = b1[nt * 16 + n];
#pragma unroll
        for (int mt = 0; mt < 2; ++mt) {
            v8f acc = __builtin_amdgcn_wmma_f32_16x16x32_f16(false, A1[mt][0], false, Bc0,
                                                             (short)0, zero, false, false);
            acc = __builtin_amdgcn_wmma_f32_16x16x32_f16(false, A1[mt][1], false, Bc1,
                                                         (short)0, acc, false, false);
#pragma unroll
            for (int r = 0; r < 8; ++r) {
                float v = fmaxf(acc[r] + bias, 0.0f);
                int pt = mt * 16 + r + 8 * hi;
                bufA[pt * 64 + nt * 16 + n] = (_Float16)v;   // act2 reuses bufA
            }
        }
    }

    // ---- layer 2: [16x64] x [64x16] (cols 3..15 zero-padded), no relu ----
    float bias2 = (n < 3) ? b2[n] : 0.0f;
    v16h B2c0 = loadB(sB2, 0, lane);
    v16h B2c1 = loadB(sB2, 1, lane);
#pragma unroll
    for (int mt = 0; mt < 2; ++mt) {
        v16h Ac0 = loadA(bufA + (mt * 16 + n) * 64, hi);
        v16h Ac1 = loadA(bufA + (mt * 16 + n) * 64 + 32, hi);
        v8f acc = __builtin_amdgcn_wmma_f32_16x16x32_f16(false, Ac0, false, B2c0,
                                                         (short)0, zero, false, false);
        acc = __builtin_amdgcn_wmma_f32_16x16x32_f16(false, Ac1, false, B2c1,
                                                     (short)0, acc, false, false);
        if (n < 3) {
#pragma unroll
            for (int r = 0; r < 8; ++r) {
                unsigned row = pbase + (unsigned)(mt * 16 + r + 8 * hi);
                out[(size_t)row * 3u + (unsigned)n] = acc[r] + bias2;
            }
        }
    }
}

extern "C" void kernel_launch(void* const* d_in, const int* in_sizes, int n_in,
                              void* d_out, int out_size, void* d_ws, size_t ws_size,
                              hipStream_t stream) {
    (void)in_sizes; (void)n_in; (void)d_ws; (void)ws_size; (void)out_size;

    const float* x    = (const float*)d_in[0];
    const float* cmax = (const float*)d_in[1];
    const float* emb  = (const float*)d_in[2];
    const float* w0   = (const float*)d_in[3];
    const float* b0   = (const float*)d_in[4];
    const float* w1   = (const float*)d_in[5];
    const float* b1   = (const float*)d_in[6];
    const float* w2   = (const float*)d_in[7];
    const float* b2   = (const float*)d_in[8];

    // Mirror _level_meta() exactly in double precision (torch-ngp geometry).
    LevelMeta meta;
    const double S = log2(2048.0 / 16.0) / 15.0;
    unsigned off = 0;
    for (int l = 0; l < 16; ++l) {
        double s = pow(2.0, (double)l * S) * 16.0 - 1.0;
        long long r = (long long)ceil(s) + 1;
        long long pc = (r + 1) * (r + 1);
        if (pc > (1ll << 24)) pc = (1ll << 24);
        pc = ((pc + 7) / 8) * 8;                 // align to 8
        meta.scale[l] = (float)s;
        meta.rs[l]    = (unsigned)(r + 1);
        meta.off[l]   = off;
        off += (unsigned)pc;
    }

    fused_hashgrid_mlp<<<NBLOCKS, BLOCK_T, 0, stream>>>(
        x, cmax, emb, w0, b0, w1, b1, w2, b2, (float*)d_out, meta);
}
